// GCN_72619307041133
// MI455X (gfx1250) — compile-verified, matrix operand order checked by reading, not run on hardware
//
#include <hip/hip_runtime.h>

// ---------------------------------------------------------------------------
// GCN 2-layer forward for gfx1250 (MI455X).
//  - Dense node GEMMs: v_wmma_f32_16x16x32_f16 (f16 in / f32 acc), W^T staged
//    in LDS via async global->LDS copy (ASYNCcnt) with 16B row pad.
//  - Edge aggregation: device-built CSR + gather-based SpMM (one wave per
//    destination node, __shfl edge broadcast) -- no feature atomics.
// ---------------------------------------------------------------------------

typedef __attribute__((ext_vector_type(16))) _Float16 v16h;
typedef __attribute__((ext_vector_type(8)))  _Float16 v8h;
typedef __attribute__((ext_vector_type(8)))  float    v8f;
typedef __attribute__((ext_vector_type(4)))  int      v4i;

#define AS1 __attribute__((address_space(1)))
#define AS3 __attribute__((address_space(3)))

#if __has_builtin(__builtin_amdgcn_global_load_async_to_lds_b128) && \
    __has_builtin(__builtin_amdgcn_s_wait_asynccnt)
#define HAVE_ASYNC_LDS 1
#endif

// ---------------- CSR build: histogram ----------------

__global__ __launch_bounds__(256) void k_zero_int(int* __restrict__ p, long n) {
  long t = (long)blockIdx.x * 256 + threadIdx.x;
  if (t < n) p[t] = 0;
}

__global__ __launch_bounds__(256) void k_count(const int* __restrict__ col,
                                               int* __restrict__ cnt, long e) {
  long t = (long)blockIdx.x * 256 + threadIdx.x;
  if (t < e) atomicAdd(&cnt[col[t]], 1);
}

__global__ __launch_bounds__(256) void k_dinv_cnt(const int* __restrict__ cnt,
                                                  float* __restrict__ dinv, long n) {
  long t = (long)blockIdx.x * 256 + threadIdx.x;
  if (t < n) dinv[t] = rsqrtf((float)cnt[t] + 1.0f);  // +1 = self-loop
}

// ---------------- CSR build: 3-stage exclusive scan (chunk = 256) ----------

__global__ __launch_bounds__(256) void k_chunk_sum(const int* __restrict__ cnt,
                                                   int* __restrict__ csum, long n) {
  __shared__ int s[256];
  long i = (long)blockIdx.x * 256 + threadIdx.x;
  s[threadIdx.x] = (i < n) ? cnt[i] : 0;
  __syncthreads();
  for (int off = 128; off > 0; off >>= 1) {
    if ((int)threadIdx.x < off) s[threadIdx.x] += s[threadIdx.x + off];
    __syncthreads();
  }
  if (threadIdx.x == 0) csum[blockIdx.x] = s[0];
}

__global__ void k_top_scan(int* __restrict__ csum, int* __restrict__ rowstart,
                           int nchunks, long n) {
  if (threadIdx.x != 0 || blockIdx.x != 0) return;
  int run = 0;
  for (int b = 0; b < nchunks; ++b) { int t = csum[b]; csum[b] = run; run += t; }
  rowstart[n] = run;  // == E
}

__global__ __launch_bounds__(256) void k_scan_emit(const int* __restrict__ cnt,
                                                   const int* __restrict__ csum,
                                                   int* __restrict__ rowstart,
                                                   int* __restrict__ cursor, long n) {
  __shared__ int s[256];
  long i = (long)blockIdx.x * 256 + threadIdx.x;
  int v = (i < n) ? cnt[i] : 0;
  s[threadIdx.x] = v;
  __syncthreads();
  for (int off = 1; off < 256; off <<= 1) {
    int x = ((int)threadIdx.x >= off) ? s[threadIdx.x - off] : 0;
    __syncthreads();
    s[threadIdx.x] += x;
    __syncthreads();
  }
  if (i < n) {
    int ex = csum[blockIdx.x] + s[threadIdx.x] - v;  // exclusive prefix
    rowstart[i] = ex;
    cursor[i]   = ex;
  }
}

__global__ __launch_bounds__(256) void k_fill(const int* __restrict__ row,
                                              const int* __restrict__ col,
                                              int* __restrict__ cursor,
                                              int* __restrict__ esrc, long e) {
  long t = (long)blockIdx.x * 256 + threadIdx.x;
  if (t < e) {
    int pos = atomicAdd(&cursor[col[t]], 1);
    esrc[pos] = row[t];
  }
}

// ---------------- weight transpose + f16 convert ----------------
// Wt[j*128 + k] = (f16) W[k*J + j]   (K = 128 for both layers)
__global__ __launch_bounds__(256) void k_wt(const float* __restrict__ W,
                                            _Float16* __restrict__ Wt, int J) {
  int t = blockIdx.x * 256 + threadIdx.x;
  if (t >= 128 * J) return;
  int j = t >> 7, k = t & 127;
  Wt[t] = (_Float16)W[(long)k * J + j];
}

// ---------------- h0 = concat(x, comm_emb[comm_ids]) in f16 ----------------
__global__ __launch_bounds__(256) void k_h0(const float* __restrict__ x,
                                            const int* __restrict__ cid,
                                            const float* __restrict__ cemb,
                                            _Float16* __restrict__ h0f,
                                            long n, int din, int cd) {
  long t = (long)blockIdx.x * 256 + threadIdx.x;
  if (t >= n * 128) return;
  long i = t >> 7;
  int  k = (int)(t & 127);
  float v = (k < din) ? x[i * din + k] : cemb[(long)cid[i] * cd + (k - din)];
  h0f[t] = (_Float16)v;
}

// ---------------- WMMA GEMM: [N,128](f16) x [128,NCOL](f16) -> [N,NCOL](f32)
template <int NCOL>
__global__ __launch_bounds__(256) void k_gemm(const _Float16* __restrict__ Hf,
                                              const _Float16* __restrict__ Wt,
                                              float* __restrict__ Z, int ntiles) {
  constexpr int LSTR = 136;                 // padded row stride in halves (272B)
  __shared__ uint4 lWs[NCOL * 17];          // 17 x uint4 = 272B per row

  const uint4* src = (const uint4*)Wt;
#if defined(HAVE_ASYNC_LDS)
  for (int ci = threadIdx.x; ci < NCOL * 16; ci += 256) {
    int j = ci >> 4, cp = ci & 15;
    __builtin_amdgcn_global_load_async_to_lds_b128(
        (AS1 v4i*)(src + ci), (AS3 v4i*)(lWs + j * 17 + cp), 0, 0);
  }
  __builtin_amdgcn_s_wait_asynccnt(0);
#else
  for (int ci = threadIdx.x; ci < NCOL * 16; ci += 256) {
    int j = ci >> 4, cp = ci & 15;
    lWs[j * 17 + cp] = src[ci];
  }
#endif
  __syncthreads();
  const _Float16* lW = (const _Float16*)lWs;

  const int wave = threadIdx.x >> 5, lane = threadIdx.x & 31;
  const int tile = blockIdx.x * 8 + wave;
  if (tile >= ntiles) return;
  const long node0 = (long)tile * 16;
  const int mn = lane & 15;   // A row / B col within tile
  const int g  = lane >> 4;   // K-half selector (ISA 16-bit A layout)

  v16h A[4];
  const _Float16* ap = Hf + (node0 + mn) * 128 + g * 8;
#pragma unroll
  for (int kt = 0; kt < 4; ++kt) {
    v8h lo = *(const v8h*)(ap + kt * 32);
    v8h hi = *(const v8h*)(ap + kt * 32 + 16);
    A[kt] = __builtin_shufflevector(lo, hi, 0, 1, 2, 3, 4, 5, 6, 7,
                                    8, 9, 10, 11, 12, 13, 14, 15);
  }

#pragma unroll
  for (int jt = 0; jt < NCOL / 16; ++jt) {
    v8f acc = {};
    const _Float16* bp = lW + (jt * 16 + mn) * LSTR + g * 8;
#pragma unroll
    for (int kt = 0; kt < 4; ++kt) {
      v8h lo = *(const v8h*)(bp + kt * 32);
      v8h hi = *(const v8h*)(bp + kt * 32 + 16);
      v16h B = __builtin_shufflevector(lo, hi, 0, 1, 2, 3, 4, 5, 6, 7,
                                       8, 9, 10, 11, 12, 13, 14, 15);
      acc = __builtin_amdgcn_wmma_f32_16x16x32_f16(
          false, A[kt], false, B, (short)0, acc, false, false);
    }
    float* zb = Z + node0 * NCOL + (jt * 16 + mn);
#pragma unroll
    for (int v = 0; v < 8; ++v) zb[(long)(v + 8 * g) * NCOL] = acc[v];
  }
}

// ---------------- gather-based aggregation: one wave per destination node ---
// out[i] = dinv[i] * sum_{e: col==i} dinv[row_e] * Z[row_e]
//        + dinv[i]^2 * Z[i] + bias   (optional ReLU, OutT in {f32, f16})
template <int NC, bool RELU, typename OutT>
__global__ __launch_bounds__(256) void k_aggr(const int* __restrict__ rowstart,
                                              const int* __restrict__ esrc,
                                              const float* __restrict__ dinv,
                                              const float* __restrict__ Z,
                                              const float* __restrict__ bias,
                                              OutT* __restrict__ out, long n) {
  long wv = ((long)blockIdx.x * 256 + threadIdx.x) >> 5;  // destination node
  int lane = threadIdx.x & 31;
  if (wv >= n) return;
  int s0 = rowstart[wv], s1 = rowstart[wv + 1];

  float a0 = 0.f, a1 = 0.f, a2 = 0.f, a3 = 0.f;
  for (int base = s0; base < s1; base += 32) {
    int idx = base + lane;
    int src = 0; float w = 0.f;
    if (idx < s1) {
      src = esrc[idx];
      w   = dinv[src];
      __builtin_prefetch(Z + (long)src * NC, 0, 0);  // global_prefetch_b8
    }
    int cnt = min(32, s1 - base);
    for (int j = 0; j < cnt; ++j) {
      int   r  = __shfl(src, j, 32);
      float ww = __shfl(w,   j, 32);
      if (NC == 128) {
        float4 v = ((const float4*)(Z + (long)r * 128))[lane];
        a0 += ww * v.x; a1 += ww * v.y; a2 += ww * v.z; a3 += ww * v.w;
      } else {
        float2 v = ((const float2*)(Z + (long)r * 64))[lane];
        a0 += ww * v.x; a1 += ww * v.y;
      }
    }
  }

  float di = dinv[wv];
  if (NC == 128) {
    float4 zs = ((const float4*)(Z + wv * 128))[lane];
    float4 bb = ((const float4*)bias)[lane];
    float r0 = di * a0 + di * di * zs.x + bb.x;
    float r1 = di * a1 + di * di * zs.y + bb.y;
    float r2 = di * a2 + di * di * zs.z + bb.z;
    float r3 = di * a3 + di * di * zs.w + bb.w;
    if (RELU) {
      r0 = r0 > 0.f ? r0 : 0.f; r1 = r1 > 0.f ? r1 : 0.f;
      r2 = r2 > 0.f ? r2 : 0.f; r3 = r3 > 0.f ? r3 : 0.f;
    }
    long o = wv * 128 + lane * 4;
    out[o + 0] = (OutT)r0; out[o + 1] = (OutT)r1;
    out[o + 2] = (OutT)r2; out[o + 3] = (OutT)r3;
  } else {
    float2 zs = ((const float2*)(Z + wv * 64))[lane];
    float2 bb = ((const float2*)bias)[lane];
    float r0 = di * a0 + di * di * zs.x + bb.x;
    float r1 = di * a1 + di * di * zs.y + bb.y;
    if (RELU) { r0 = r0 > 0.f ? r0 : 0.f; r1 = r1 > 0.f ? r1 : 0.f; }
    long o = wv * 64 + lane * 2;
    out[o + 0] = (OutT)r0; out[o + 1] = (OutT)r1;
  }
}

// ---------------------------------------------------------------------------

extern "C" void kernel_launch(void* const* d_in, const int* in_sizes, int n_in,
                              void* d_out, int out_size, void* d_ws, size_t ws_size,
                              hipStream_t stream) {
  const float* x    = (const float*)d_in[0];
  const int*   ei   = (const int*)d_in[1];   // edge_index [2,E]
  const int*   cid  = (const int*)d_in[2];
  const float* cemb = (const float*)d_in[3];
  const float* W1   = (const float*)d_in[4];
  const float* b1   = (const float*)d_in[5];
  const float* W2   = (const float*)d_in[6];
  const float* b2   = (const float*)d_in[7];
  float* out = (float*)d_out;

  const long N  = in_sizes[2];
  const long E  = in_sizes[1] / 2;
  const int din = (int)(in_sizes[0] / N);    // 120
  const int cd  = in_sizes[3] / 64;          // 8 ; din + cd == 128
  const int* row = ei;
  const int* col = ei + E;

  // --- workspace carve (all 256B aligned) ---
  char* ws = (char*)d_ws;
  auto carve = [&](size_t bytes) {
    char* p = ws;
    ws += (bytes + 255) & ~(size_t)255;
    return p;
  };
  const int nchunks = (int)((N + 255) / 256);
  float*    dinv     = (float*)carve(N * 4);
  int*      cnti     = (int*)carve(N * 4);
  int*      rowstart = (int*)carve((N + 1) * 4);
  int*      cursor   = (int*)carve(N * 4);
  int*      esrc     = (int*)carve(E * 4);
  int*      csum     = (int*)carve((size_t)nchunks * 4);
  _Float16* h0f      = (_Float16*)carve(N * 128 * 2);
  _Float16* w1t      = (_Float16*)carve(128 * 128 * 2);
  _Float16* w2t      = (_Float16*)carve(64 * 128 * 2);
  float*    z1       = (float*)carve(N * 128 * 4);
  _Float16* h1f      = (_Float16*)carve(N * 128 * 2);
  float*    z2       = (float*)carve(N * 64 * 4);

  const int BN    = (int)((N + 255) / 256);
  const int BE    = (int)((E + 255) / 256);
  const int BN128 = (int)((N * 128 + 255) / 256);
  const int ntiles = (int)(N / 16);               // N = 100000 -> 6250
  const int Bgemm  = (ntiles + 7) / 8;
  const int Baggr  = (int)((N * 32 + 255) / 256); // one wave per node

  // --- CSR build + symmetric norm (shared by both layers) ---
  k_zero_int<<<BN, 256, 0, stream>>>(cnti, N);
  k_count<<<BE, 256, 0, stream>>>(col, cnti, E);
  k_dinv_cnt<<<BN, 256, 0, stream>>>(cnti, dinv, N);
  k_chunk_sum<<<nchunks, 256, 0, stream>>>(cnti, csum, N);
  k_top_scan<<<1, 1, 0, stream>>>(csum, rowstart, nchunks, N);
  k_scan_emit<<<nchunks, 256, 0, stream>>>(cnti, csum, rowstart, cursor, N);
  k_fill<<<BE, 256, 0, stream>>>(row, col, cursor, esrc, E);

  // --- weights -> f16 transposed ---
  k_wt<<<(128 * 128 + 255) / 256, 256, 0, stream>>>(W1, w1t, 128);
  k_wt<<<(128 * 64 + 255) / 256, 256, 0, stream>>>(W2, w2t, 64);

  // --- layer 1 ---
  k_h0<<<BN128, 256, 0, stream>>>(x, cid, cemb, h0f, N, din, cd);
  k_gemm<128><<<Bgemm, 256, 0, stream>>>(h0f, w1t, z1, ntiles);
  k_aggr<128, true, _Float16><<<Baggr, 256, 0, stream>>>(rowstart, esrc, dinv,
                                                         z1, b1, h1f, N);
  // --- layer 2 ---
  k_gemm<64><<<Bgemm, 256, 0, stream>>>(h1f, w2t, z2, ntiles);
  k_aggr<64, false, float><<<Baggr, 256, 0, stream>>>(rowstart, esrc, dinv,
                                                      z2, b2, out, N);
}
